// BCNN_69870527971725
// MI455X (gfx1250) — 8-bit matrix operand order repaired and checked by reading, compiled, not run on hardware
//
#include <hip/hip_runtime.h>
#include <stdint.h>

typedef __attribute__((ext_vector_type(8))) int v8i;

#define BN_EPS 1e-3f

__device__ __forceinline__ int imax(int a, int b) { return a > b ? a : b; }

__global__ void zeropage_kernel(int* __restrict__ zp) { zp[threadIdx.x] = 0; }

__global__ void wprep_kernel(const float* __restrict__ w, int8_t* __restrict__ wb,
                             int Cin, int Cout, int Kpad) {
    int idx = blockIdx.x * blockDim.x + threadIdx.x;
    int total = Cout * Kpad;
    if (idx >= total) return;
    int cout = idx / Kpad;
    int k    = idx - cout * Kpad;
    int Ktrue = 9 * Cin;
    int8_t v = 0;
    if (k < Ktrue) {
        int tap = k / Cin;
        int cin = k - tap * Cin;
        float f = w[(tap * Cin + cin) * Cout + cout];
        v = (f >= 0.f) ? (int8_t)1 : (int8_t)-1;
    }
    wb[idx] = v;
}

__global__ __launch_bounds__(256) void stem_kernel(
        const float* __restrict__ x, const float* __restrict__ w1,
        const float* __restrict__ m1, const float* __restrict__ v1,
        const float* __restrict__ b1, int8_t* __restrict__ act1) {
    __shared__ float ws[3 * 3 * 3 * 32];
    for (int i = threadIdx.x; i < 864; i += blockDim.x)
        ws[i] = (w1[i] >= 0.f) ? 1.f : -1.f;
    __syncthreads();

    const int PH = 63, PW = 63, CO = 32, H = 128, W = 128, CI = 3;
    int idx = blockIdx.x * blockDim.x + threadIdx.x;
    if (idx >= 64 * PH * PW) return;
    int pw = idx % PW;
    int t = idx / PW;
    int ph = t % PH;
    int b  = t / PH;

    float patch[4][4][3];
    int oh0 = 2 * ph, ow0 = 2 * pw;
    #pragma unroll
    for (int r = 0; r < 4; ++r) {
        const float* xr = x + ((size_t)(b * H + oh0 + r) * W + ow0) * CI;
        #pragma unroll
        for (int cc = 0; cc < 4; ++cc)
            #pragma unroll
            for (int ci = 0; ci < 3; ++ci)
                patch[r][cc][ci] = xr[cc * CI + ci];
    }

    int8_t* dst = act1 + (size_t)idx * CO;
    for (int cout = 0; cout < CO; ++cout) {
        float a00 = 0.f, a01 = 0.f, a10 = 0.f, a11 = 0.f;
        #pragma unroll
        for (int kh = 0; kh < 3; ++kh)
            #pragma unroll
            for (int kw = 0; kw < 3; ++kw)
                #pragma unroll
                for (int ci = 0; ci < 3; ++ci) {
                    float wv = ws[((kh * 3 + kw) * 3 + ci) * CO + cout];
                    a00 = fmaf(patch[kh][kw][ci],         wv, a00);
                    a01 = fmaf(patch[kh][kw + 1][ci],     wv, a01);
                    a10 = fmaf(patch[kh + 1][kw][ci],     wv, a10);
                    a11 = fmaf(patch[kh + 1][kw + 1][ci], wv, a11);
                }
        float mx = fmaxf(fmaxf(a00, a01), fmaxf(a10, a11));
        float bn = (mx - m1[cout]) * rsqrtf(v1[cout] + BN_EPS) + b1[cout];
        dst[cout] = (bn >= 0.f) ? (int8_t)1 : (int8_t)-1;
    }
}

template<int H, int W, int Cin, int Cout, int PoolH, int PoolW, int ColGroups,
         int NChunks, bool Last>
__global__ __launch_bounds__(128) void binconv_kernel(
        const int8_t* __restrict__ act,
        const int8_t* __restrict__ wb,
        const float* __restrict__ bn_m, const float* __restrict__ bn_v,
        const float* __restrict__ bn_b, const int8_t* __restrict__ zp,
        int8_t* __restrict__ actOut, float* __restrict__ fOut)
{
    constexpr int Kpad  = NChunks * 64;
    constexpr int Ktrue = 9 * Cin;

    int lane = threadIdx.x & 31;
    int wave = threadIdx.x >> 5;
    int prr  = blockIdx.x / ColGroups;
    int g    = blockIdx.x - prr * ColGroups;
    int b    = blockIdx.y;
    int l16  = lane & 15;
    int half = lane >> 4;
    int cout = (blockIdx.z * 4 + wave) * 16 + l16;

    const int8_t* actB = act + (size_t)b * H * W * Cin;

    int w  = 8 * g + (l16 & 7);
    int rb = l16 >> 3;
    int h0 = 4 * prr + rb;
    int h1 = h0 + 2;
    int hOff = 8 * half;

    const int8_t* tp0[9];
    const int8_t* tp1[9];
    #pragma unroll
    for (int t = 0; t < 9; ++t) {
        int dh = t / 3 - 1, dw = t % 3 - 1;
        int iw  = w + dw;
        int ih0 = h0 + dh, ih1 = h1 + dh;
        bool colOk = (iw >= 0) && (iw < W) && (w < W);
        bool v0 = colOk && (ih0 >= 0) && (ih0 < H) && (h0 < H);
        bool v1 = colOk && (ih1 >= 0) && (ih1 < H) && (h1 < H);
        tp0[t] = (v0 ? actB + (size_t)(ih0 * W + iw) * Cin : zp) + hOff;
        tp1[t] = (v1 ? actB + (size_t)(ih1 * W + iw) * Cin : zp) + hOff;
    }
    const int8_t* wRow = wb + (size_t)cout * Kpad + 8 * half;
    __builtin_prefetch(wRow, 0, 3);

    v8i c0 = {}, c1 = {};
    #pragma unroll
    for (int ch = 0; ch < NChunks; ++ch) {
        v8i a0, a1, bf;
        #pragma unroll
        for (int p = 0; p < 4; ++p) {
            int kBlk = ch * 64 + 16 * p;
            if (kBlk < Ktrue) {
                int t    = kBlk / Cin;
                int cinB = kBlk % Cin;
                a0[2 * p]     = *(const int*)(tp0[t] + cinB);
                a0[2 * p + 1] = *(const int*)(tp0[t] + cinB + 4);
                a1[2 * p]     = *(const int*)(tp1[t] + cinB);
                a1[2 * p + 1] = *(const int*)(tp1[t] + cinB + 4);
            } else {
                a0[2 * p] = 0; a0[2 * p + 1] = 0;
                a1[2 * p] = 0; a1[2 * p + 1] = 0;
            }
        }
        #pragma unroll
        for (int r = 0; r < 8; ++r)
            bf[r] = *(const int*)(wRow + ch * 64 + 16 * (r >> 1) + 4 * (r & 1));
        c0 = __builtin_amdgcn_wmma_i32_16x16x64_iu8(true, a0, true, bf, c0, false, false);
        c1 = __builtin_amdgcn_wmma_i32_16x16x64_iu8(true, a1, true, bf, c1, false, false);
    }

    float rs   = rsqrtf(bn_v[cout] + BN_EPS);
    float mean = bn_m[cout];
    float beta = bn_b[cout];

    #pragma unroll
    for (int reg = 0; reg < 2; ++reg) {
        v8i& c = reg ? c1 : c0;
        int pr = 2 * prr + reg;
        if (pr < PoolH) {
            int p0 = imax(c[0], c[1]);
            int p1 = imax(c[2], c[3]);
            int p2 = imax(c[4], c[5]);
            int p3 = imax(c[6], c[7]);
            p0 = imax(p0, __shfl_xor(p0, 16, 32));
            p1 = imax(p1, __shfl_xor(p1, 16, 32));
            p2 = imax(p2, __shfl_xor(p2, 16, 32));
            p3 = imax(p3, __shfl_xor(p3, 16, 32));
            int va = half ? p2 : p0;
            int vb = half ? p3 : p1;
            int pa = half ? 2 : 0;
            #pragma unroll
            for (int q = 0; q < 2; ++q) {
                int pc = 4 * g + pa + q;
                if (pc < PoolW) {
                    float bn = ((float)(q ? vb : va) - mean) * rs + beta;
                    size_t o = (((size_t)b * PoolH + pr) * PoolW + pc) * Cout + cout;
                    if (Last) fOut[o] = bn;
                    else      actOut[o] = (bn >= 0.f) ? (int8_t)1 : (int8_t)-1;
                }
            }
        }
    }
}

extern "C" void kernel_launch(void* const* d_in, const int* in_sizes, int n_in,
                              void* d_out, int out_size, void* d_ws, size_t ws_size,
                              hipStream_t stream) {
    const float* x  = (const float*)d_in[0];
    const float* w1 = (const float*)d_in[1];
    const float* m1 = (const float*)d_in[2];
    const float* v1 = (const float*)d_in[3];
    const float* b1 = (const float*)d_in[4];
    const float* w2 = (const float*)d_in[5];
    const float* m2 = (const float*)d_in[6];
    const float* v2 = (const float*)d_in[7];
    const float* b2 = (const float*)d_in[8];
    const float* w3 = (const float*)d_in[9];
    const float* m3 = (const float*)d_in[10];
    const float* v3 = (const float*)d_in[11];
    const float* b3 = (const float*)d_in[12];
    const float* w4 = (const float*)d_in[13];
    const float* m4 = (const float*)d_in[14];
    const float* v4 = (const float*)d_in[15];
    const float* b4 = (const float*)d_in[16];
    float* out = (float*)d_out;

    uint8_t* wsp = (uint8_t*)d_ws;
    size_t off = 0;
    auto alloc = [&](size_t bytes) {
        uint8_t* p = wsp + off;
        off = (off + bytes + 255) & ~(size_t)255;
        return p;
    };
    int8_t* zp   = (int8_t*)alloc(256);
    int8_t* act1 = (int8_t*)alloc((size_t)64 * 63 * 63 * 32);
    int8_t* act2 = (int8_t*)alloc((size_t)64 * 31 * 31 * 64);
    int8_t* act3 = (int8_t*)alloc((size_t)64 * 15 * 15 * 128);
    int8_t* wb2  = (int8_t*)alloc((size_t)64 * 320);
    int8_t* wb3  = (int8_t*)alloc((size_t)128 * 576);
    int8_t* wb4  = (int8_t*)alloc((size_t)256 * 1152);
    (void)ws_size; (void)in_sizes; (void)n_in; (void)out_size;

    zeropage_kernel<<<1, 64, 0, stream>>>((int*)zp);

    wprep_kernel<<<(64 * 320 + 255) / 256, 256, 0, stream>>>(w2, wb2, 32, 64, 320);
    wprep_kernel<<<(128 * 576 + 255) / 256, 256, 0, stream>>>(w3, wb3, 64, 128, 576);
    wprep_kernel<<<(256 * 1152 + 255) / 256, 256, 0, stream>>>(w4, wb4, 128, 256, 1152);

    {
        int total = 64 * 63 * 63;
        stem_kernel<<<(total + 255) / 256, 256, 0, stream>>>(x, w1, m1, v1, b1, act1);
    }

    binconv_kernel<63, 63, 32, 64, 31, 31, 8, 5, false>
        <<<dim3(16 * 8, 64, 1), 128, 0, stream>>>(
            act1, wb2, m2, v2, b2, zp, act2, nullptr);

    binconv_kernel<31, 31, 64, 128, 15, 15, 4, 9, false>
        <<<dim3(8 * 4, 64, 2), 128, 0, stream>>>(
            act2, wb3, m3, v3, b3, zp, act3, nullptr);

    binconv_kernel<15, 15, 128, 256, 7, 7, 2, 18, true>
        <<<dim3(4 * 2, 64, 4), 128, 0, stream>>>(
            act3, wb4, m4, v4, b4, zp, nullptr, out);
}
